// SelfAttention2D_11166914970565
// MI455X (gfx1250) — matrix operand order repaired and checked by reading, not compiled, not run on hardware
//
#include <hip/hip_runtime.h>

// ---------------------------------------------------------------------------
// MI455X (gfx1250) bf16-WMMA implementation of SelfAttention2D.
// ~103 GFLOP total, ~1 GB min HBM traffic -> bandwidth-limited (~43us @23.3TB/s)
// if compute runs at bf16 WMMA rates; hence v_wmma_f32_16x16x32_bf16 everywhere.
// ---------------------------------------------------------------------------

typedef unsigned short u16;
typedef __attribute__((ext_vector_type(16))) __bf16 v16bf;
typedef __attribute__((ext_vector_type(8)))  float  v8f;
typedef __attribute__((ext_vector_type(16))) u16    u16x16;
typedef __attribute__((ext_vector_type(8)))  u16    u16x8;
typedef __attribute__((ext_vector_type(4)))  float  f32x4;

union V16 { v16bf bf; u16x16 us; u16x8 h[2]; };

#define CIN   256
#define DH    128
#define REGUL 0.08838834764831845f   // sqrt(2/256)
#define BNEPS 1e-3f

__device__ __forceinline__ u16 f2bf(float f) {           // RNE f32 -> bf16
  unsigned u = __float_as_uint(f);
  u += 0x7FFFu + ((u >> 16) & 1u);
  return (u16)(u >> 16);
}

__device__ __forceinline__ v8f wmma_bf16(const V16 a, const V16 b, v8f c) {
  return __builtin_amdgcn_wmma_f32_16x16x32_bf16(false, a.bf, false, b.bf,
                                                 (short)0, c, false, false);
}

// A-fragment (16x32 bf16): lane L, elem e -> M = L&15, K = 16*(e>>3)+8*(L>>4)+(e&7)
__device__ __forceinline__ V16 load_a(const u16* base, int stride, int row,
                                      int kbase, int hsel) {
  V16 r;
  const u16* p = base + row * stride + kbase + 8 * hsel;
  r.h[0] = *(const u16x8*)p;
  r.h[1] = *(const u16x8*)(p + 16);
  return r;
}

// B-fragment (32x16 bf16): lane L, elem e -> N = L&15, K = 16*(e>>3)+8*(L>>4)+(e&7)
// base is stored "transposed" [N][K] row-major -> two 16B chunks per lane.
__device__ __forceinline__ V16 load_b(const u16* base, int stride, int row,
                                      int kbase, int hsel) {
  V16 r;
  const u16* p = base + row * stride + kbase + 8 * hsel;
  r.h[0] = *(const u16x8*)p;
  r.h[1] = *(const u16x8*)(p + 16);
  return r;
}

// ---------------------------------------------------------------------------
// Kernel 0: fold BN into conv weight, transpose to [D][K] bf16, emit bias.
// ---------------------------------------------------------------------------
__global__ void fold_weights(const float* __restrict__ W, const float* __restrict__ g,
                             const float* __restrict__ b, const float* __restrict__ m,
                             const float* __restrict__ v, u16* __restrict__ Wt,
                             float* __restrict__ bias, int K, int D) {
  int idx = blockIdx.x * blockDim.x + threadIdx.x;
  if (idx < K * D) {
    int k = idx / D, d = idx % D;
    float s = 1.0f;
    if (g) s = g[d] * rsqrtf(v[d] + BNEPS);
    Wt[d * K + k] = f2bf(W[k * D + d] * s);
  }
  if (bias != nullptr && idx < D) {
    float s = g[idx] * rsqrtf(v[idx] + BNEPS);
    bias[idx] = b[idx] - m[idx] * s;
  }
}

// ---------------------------------------------------------------------------
// Kernel 1: per-pixel GEMM chain. Block = 128 pixels, 8 waves x 16 pixels.
// LDS: x-tile bf16 [128][256] (64KB) + per-wave h staging [8][16][128] (32KB).
// ---------------------------------------------------------------------------
__device__ __forceinline__ void branch_pass(
    const u16* xt, u16* hst_w,
    const u16* __restrict__ w1, const float* __restrict__ b1,
    const u16* __restrict__ w2, const float* __restrict__ b2,
    u16* __restrict__ out, long pix0, int wave, int mrow, int hsel)
{
  // GEMM1: [16 x 256] @ [256 x 128] + bias + relu
  v8f acc[8] = {};
  #pragma unroll
  for (int k = 0; k < 8; ++k) {
    V16 a = load_a(xt, CIN, wave * 16 + mrow, k * 32, hsel);
    #pragma unroll
    for (int n = 0; n < 8; ++n) {
      V16 b = load_b(w1, CIN, n * 16 + mrow, k * 32, hsel);
      acc[n] = wmma_bf16(a, b, acc[n]);
    }
  }
  #pragma unroll
  for (int n = 0; n < 8; ++n) {
    float bb = b1[n * 16 + mrow];
    #pragma unroll
    for (int r = 0; r < 8; ++r)
      hst_w[(r + 8 * hsel) * DH + n * 16 + mrow] = f2bf(fmaxf(acc[n][r] + bb, 0.f));
  }
  // GEMM2: [16 x 128] @ [128 x 128] + bias + relu -> global bf16
  v8f acc2[8] = {};
  #pragma unroll
  for (int k = 0; k < 4; ++k) {
    V16 a = load_a(hst_w, DH, mrow, k * 32, hsel);
    #pragma unroll
    for (int n = 0; n < 8; ++n) {
      V16 b = load_b(w2, DH, n * 16 + mrow, k * 32, hsel);
      acc2[n] = wmma_bf16(a, b, acc2[n]);
    }
  }
  #pragma unroll
  for (int n = 0; n < 8; ++n) {
    float bb = b2[n * 16 + mrow];
    #pragma unroll
    for (int r = 0; r < 8; ++r)
      out[(pix0 + wave * 16 + r + 8 * hsel) * DH + n * 16 + mrow] =
          f2bf(fmaxf(acc2[n][r] + bb, 0.f));
  }
}

__global__ __launch_bounds__(256, 1) void branch_kernel(
    const float* __restrict__ x,
    const u16* __restrict__ wT1, const float* __restrict__ bT1,
    const u16* __restrict__ wT2, const float* __restrict__ bT2,
    const u16* __restrict__ wP1, const float* __restrict__ bP1,
    const u16* __restrict__ wP2, const float* __restrict__ bP2,
    const u16* __restrict__ wG,
    u16* __restrict__ theta, u16* __restrict__ phi, u16* __restrict__ gamma)
{
  extern __shared__ __align__(16) char smem[];
  u16* xt  = (u16*)smem;                    // [128][256] bf16
  u16* hst = (u16*)(smem + 128 * CIN * 2);  // [8][16][128] bf16

  const int tid  = threadIdx.x;
  const int wave = tid >> 5, lane = tid & 31;
  const int mrow = lane & 15, hsel = lane >> 4;
  const long pix0 = (long)blockIdx.x * 128;

  // stage x tile -> bf16 LDS (read x exactly once per block)
  const f32x4* xg = (const f32x4*)(x + pix0 * CIN);
  #pragma unroll
  for (int i = 0; i < 32; ++i) {
    int e4 = i * 256 + tid;                 // 8192 float4 groups
    f32x4 vv = xg[e4];
    u16* dst = xt + e4 * 4;
    dst[0] = f2bf(vv.x); dst[1] = f2bf(vv.y);
    dst[2] = f2bf(vv.z); dst[3] = f2bf(vv.w);
  }
  __syncthreads();

  u16* hst_w = hst + wave * 16 * DH;
  branch_pass(xt, hst_w, wT1, bT1, wT2, bT2, theta, pix0, wave, mrow, hsel);
  branch_pass(xt, hst_w, wP1, bP1, wP2, bP2, phi,   pix0, wave, mrow, hsel);

  // gamma = x @ g_w  (no BN / relu)
  v8f acc[8] = {};
  #pragma unroll
  for (int k = 0; k < 8; ++k) {
    V16 a = load_a(xt, CIN, wave * 16 + mrow, k * 32, hsel);
    #pragma unroll
    for (int n = 0; n < 8; ++n) {
      V16 b = load_b(wG, CIN, n * 16 + mrow, k * 32, hsel);
      acc[n] = wmma_bf16(a, b, acc[n]);
    }
  }
  #pragma unroll
  for (int n = 0; n < 8; ++n)
    #pragma unroll
    for (int r = 0; r < 8; ++r)
      gamma[(pix0 + wave * 16 + r + 8 * hsel) * DH + n * 16 + mrow] = f2bf(acc[n][r]);
}

// ---------------------------------------------------------------------------
// Kernel 2: one (b,h) row per block. LDS: theta[128][128] + phi[128][128] +
// gammaT[128][128] + P-staging [8][16][128], all bf16 => 128KB (WGP has 320KB).
// ---------------------------------------------------------------------------
__global__ __launch_bounds__(256, 1) void attn_kernel(
    const u16* __restrict__ theta, const u16* __restrict__ phi,
    const u16* __restrict__ gamma, const u16* __restrict__ wR,
    float* __restrict__ out)
{
  extern __shared__ __align__(16) char smem[];
  u16* th  = (u16*)smem;            // [w][c]
  u16* ph  = th + 128 * DH;         // [v][c]
  u16* gT  = ph + 128 * DH;         // gamma transposed: [c][v]
  u16* pst = gT + 128 * DH;         // [8][16][128]

  const int tid  = threadIdx.x;
  const int wave = tid >> 5, lane = tid & 31;
  const int mrow = lane & 15, hsel = lane >> 4;
  const long pix0 = (long)blockIdx.x * 128;

  // stage theta/phi row-major, gamma transposed
  const u16x8* thg = (const u16x8*)(theta + pix0 * DH);
  const u16x8* phg = (const u16x8*)(phi + pix0 * DH);
  u16x8* thl = (u16x8*)th;
  u16x8* phl = (u16x8*)ph;
  #pragma unroll
  for (int i = 0; i < 8; ++i) {
    int e = i * 256 + tid;                  // 2048 groups of 8 bf16
    thl[e] = thg[e];
    phl[e] = phg[e];
  }
  #pragma unroll
  for (int i = 0; i < 8; ++i) {
    int e = i * 256 + tid;
    int vrow = e >> 4;                      // 16 groups per 128-col row
    int c0 = (e & 15) * 8;
    u16x8 gv = *(const u16x8*)(gamma + (pix0 + vrow) * DH + c0);
    #pragma unroll
    for (int j = 0; j < 8; ++j) gT[(c0 + j) * 128 + vrow] = gv[j];
  }
  __syncthreads();

  // S = REGUL * theta @ phi^T   (B[K=c,N=v] = phi[v][c] -> ph row-major works)
  v8f acc[8] = {};
  #pragma unroll
  for (int k = 0; k < 4; ++k) {
    V16 a = load_a(th, DH, wave * 16 + mrow, k * 32, hsel);
    #pragma unroll
    for (int n = 0; n < 8; ++n) {
      V16 b = load_b(ph, DH, n * 16 + mrow, k * 32, hsel);
      acc[n] = wmma_bf16(a, b, acc[n]);
    }
  }

  // row softmax: row M = r + 8*hsel spans lanes (mrow 0..15, same hsel) x 8 tiles
  u16* pw = pst + wave * 16 * DH;
  #pragma unroll
  for (int r = 0; r < 8; ++r) {
    float mx = -3.4e38f;
    #pragma unroll
    for (int n = 0; n < 8; ++n) { acc[n][r] *= REGUL; mx = fmaxf(mx, acc[n][r]); }
    #pragma unroll
    for (int s = 1; s <= 8; s <<= 1) mx = fmaxf(mx, __shfl_xor(mx, s, 32));
    float ex[8]; float sum = 0.f;
    #pragma unroll
    for (int n = 0; n < 8; ++n) { ex[n] = __expf(acc[n][r] - mx); sum += ex[n]; }
    #pragma unroll
    for (int s = 1; s <= 8; s <<= 1) sum += __shfl_xor(sum, s, 32);
    float inv = 1.f / sum;
    #pragma unroll
    for (int n = 0; n < 8; ++n)
      pw[(r + 8 * hsel) * DH + n * 16 + mrow] = f2bf(ex[n] * inv);
  }

  // O = P @ gamma  (B[K=v,N=c] = gamma[v][c] = gT[c][v] -> contiguous)
  v8f oacc[8] = {};
  #pragma unroll
  for (int k = 0; k < 4; ++k) {
    V16 a = load_a(pw, DH, mrow, k * 32, hsel);
    #pragma unroll
    for (int n = 0; n < 8; ++n) {
      V16 b = load_b(gT, 128, n * 16 + mrow, k * 32, hsel);
      oacc[n] = wmma_bf16(a, b, oacc[n]);
    }
  }
  // restage O (bf16) into pw for the final projection
  #pragma unroll
  for (int n = 0; n < 8; ++n)
    #pragma unroll
    for (int r = 0; r < 8; ++r)
      pw[(r + 8 * hsel) * DH + n * 16 + mrow] = f2bf(oacc[n][r]);

  // out = O @ r_w ; wR stored as [256][128] = r_w^T. N=256 in two halves.
  #pragma unroll
  for (int half = 0; half < 2; ++half) {
    v8f f[8] = {};
    #pragma unroll
    for (int k = 0; k < 4; ++k) {
      V16 a = load_a(pw, DH, mrow, k * 32, hsel);
      #pragma unroll
      for (int n = 0; n < 8; ++n) {
        V16 b = load_b(wR, DH, half * 128 + n * 16 + mrow, k * 32, hsel);
        f[n] = wmma_bf16(a, b, f[n]);
      }
    }
    #pragma unroll
    for (int n = 0; n < 8; ++n)
      #pragma unroll
      for (int r = 0; r < 8; ++r)
        out[(pix0 + wave * 16 + r + 8 * hsel) * 256 + half * 128 + n * 16 + mrow] = f[n][r];
  }
}

// ---------------------------------------------------------------------------
extern "C" void kernel_launch(void* const* d_in, const int* in_sizes, int n_in,
                              void* d_out, int out_size, void* d_ws, size_t ws_size,
                              hipStream_t stream) {
  (void)in_sizes; (void)n_in; (void)out_size; (void)ws_size;
  const float* x    = (const float*)d_in[0];
  const float* t_w1 = (const float*)d_in[1];
  const float* t1_g = (const float*)d_in[2];
  const float* t1_b = (const float*)d_in[3];
  const float* t1_m = (const float*)d_in[4];
  const float* t1_v = (const float*)d_in[5];
  const float* t_w2 = (const float*)d_in[6];
  const float* t2_g = (const float*)d_in[7];
  const float* t2_b = (const float*)d_in[8];
  const float* t2_m = (const float*)d_in[9];
  const float* t2_v = (const float*)d_in[10];
  const float* p_w1 = (const float*)d_in[11];
  const float* p1_g = (const float*)d_in[12];
  const float* p1_b = (const float*)d_in[13];
  const float* p1_m = (const float*)d_in[14];
  const float* p1_v = (const float*)d_in[15];
  const float* p_w2 = (const float*)d_in[16];
  const float* p2_g = (const float*)d_in[17];
  const float* p2_b = (const float*)d_in[18];
  const float* p2_m = (const float*)d_in[19];
  const float* p2_v = (const float*)d_in[20];
  const float* g_w  = (const float*)d_in[21];
  const float* r_w  = (const float*)d_in[22];

  char* ws = (char*)d_ws;
  size_t off = 0;
  auto take = [&](size_t bytes) -> void* {
    off = (off + 255) & ~(size_t)255;
    void* p = ws + off; off += bytes; return p;
  };
  u16*   wT1 = (u16*)  take(DH * CIN * 2);
  float* bT1 = (float*)take(DH * 4);
  u16*   wT2 = (u16*)  take(DH * DH * 2);
  float* bT2 = (float*)take(DH * 4);
  u16*   wP1 = (u16*)  take(DH * CIN * 2);
  float* bP1 = (float*)take(DH * 4);
  u16*   wP2 = (u16*)  take(DH * DH * 2);
  float* bP2 = (float*)take(DH * 4);
  u16*   wG  = (u16*)  take(DH * CIN * 2);
  u16*   wR  = (u16*)  take(CIN * DH * 2);
  const size_t NPIX = 16l * 128 * 128;            // 262144 pixels
  u16* theta = (u16*)take(NPIX * DH * 2);
  u16* phi   = (u16*)take(NPIX * DH * 2);
  u16* gamma = (u16*)take(NPIX * DH * 2);

  // fold BN + transpose weights (tiny)
  fold_weights<<<(CIN * DH + 255) / 256, 256, 0, stream>>>(t_w1, t1_g, t1_b, t1_m, t1_v, wT1, bT1, CIN, DH);
  fold_weights<<<(DH  * DH + 255) / 256, 256, 0, stream>>>(t_w2, t2_g, t2_b, t2_m, t2_v, wT2, bT2, DH, DH);
  fold_weights<<<(CIN * DH + 255) / 256, 256, 0, stream>>>(p_w1, p1_g, p1_b, p1_m, p1_v, wP1, bP1, CIN, DH);
  fold_weights<<<(DH  * DH + 255) / 256, 256, 0, stream>>>(p_w2, p2_g, p2_b, p2_m, p2_v, wP2, bP2, DH, DH);
  fold_weights<<<(CIN * DH + 255) / 256, 256, 0, stream>>>(g_w, nullptr, nullptr, nullptr, nullptr, wG, nullptr, CIN, DH);
  fold_weights<<<(DH * CIN + 255) / 256, 256, 0, stream>>>(r_w, nullptr, nullptr, nullptr, nullptr, wR, nullptr, DH, CIN);

  // per-pixel GEMM chain: 2048 blocks x 128 pixels, 96KB dynamic LDS
  branch_kernel<<<2048, 256, (128 * CIN + 8 * 16 * DH) * 2, stream>>>(
      x, wT1, bT1, wT2, bT2, wP1, bP1, wP2, bP2, wG, theta, phi, gamma);

  // attention + output projection: one (b,h) row per block, 128KB dynamic LDS
  attn_kernel<<<2048, 256, (3 * 128 * DH + 8 * 16 * DH) * 2, stream>>>(
      theta, phi, gamma, wR, (float*)d_out);
}
